// GraphAttention_68839735820715
// MI455X (gfx1250) — matrix operand order repaired and checked by reading, not compiled, not run on hardware
//
#include <hip/hip_runtime.h>
#include <hip/hip_bf16.h>
#include <stdint.h>

// GAT: N=4096 nodes, IN_DIM=512, H=4 heads, D=64. out f32 (4096,256).
#define NN   4096
#define KDIM 512
#define HEADS 4
#define DD   64
#define OC   256   // HEADS*DD

#define L2E 1.4426950408889634f

typedef __attribute__((ext_vector_type(16))) __bf16 v16bf;
typedef __attribute__((ext_vector_type(8)))  float  v8f;

union ABf { v16bf v; __bf16 b[16]; uint4 q[2]; };
union Cf  { v8f v; float f[8]; };

__device__ __forceinline__ float leaky(float x) { return fmaxf(x, 0.2f * x); }

// ---------------- K1: W (512x256 f32) -> Wt hi/lo (256x512 bf16, transposed) --------
__global__ void k_prep_w(const float* __restrict__ W,
                         __bf16* __restrict__ Wt_hi,
                         __bf16* __restrict__ Wt_lo) {
  int idx = blockIdx.x * blockDim.x + threadIdx.x;
  if (idx >= OC * KDIM) return;
  int n = idx >> 9, k = idx & (KDIM - 1);
  float f = W[k * OC + n];
  __bf16 hb = (__bf16)f;               // hardware v_cvt RNE
  Wt_hi[idx] = hb;
  Wt_lo[idx] = (__bf16)(f - (float)hb);
}

// ---------------- K2: h = X@W via bf16-split WMMA (fp32-accurate), fused s1/s2 ------
// block = 128 threads = 4 waves; wave w -> head w; block -> 16-row i-tile.
__global__ void __launch_bounds__(128) k_gemm_h(
    const float* __restrict__ X, const float* __restrict__ attn,
    const __bf16* __restrict__ Wt_hi, const __bf16* __restrict__ Wt_lo,
    __bf16* __restrict__ h_hi, __bf16* __restrict__ h_lo,
    float* __restrict__ s1t, float* __restrict__ s2t) {
  const int lane = threadIdx.x & 31;
  const int head = threadIdx.x >> 5;
  const int row  = lane & 15;
  const int lh   = lane >> 4;
  const int kb   = lh * 8;
  const int klo  = lh * 16;
  const int i0   = blockIdx.x * 16;

  Cf acc[4];
  #pragma unroll
  for (int dt = 0; dt < 4; ++dt)
    #pragma unroll
    for (int e = 0; e < 8; ++e) acc[dt].f[e] = 0.0f;

  const float* Xr = X + (size_t)(i0 + row) * KDIM;
  for (int k0 = 0; k0 < KDIM; k0 += 32) {
    // A tile in WMMA 16-bit A layout: b[0..7] = K kb+0..7, b[8..15] = K 16+kb+0..7
    float4 a01 = *(const float4*)(Xr + k0 + kb);
    float4 a23 = *(const float4*)(Xr + k0 + kb + 4);
    float4 a45 = *(const float4*)(Xr + k0 + 16 + kb);
    float4 a67 = *(const float4*)(Xr + k0 + 16 + kb + 4);
    float tmp[16] = {a01.x,a01.y,a01.z,a01.w, a23.x,a23.y,a23.z,a23.w,
                     a45.x,a45.y,a45.z,a45.w, a67.x,a67.y,a67.z,a67.w};
    ABf ahi, alo;
    #pragma unroll
    for (int q = 0; q < 16; ++q) {
      __bf16 hb = (__bf16)tmp[q];
      ahi.b[q] = hb;
      alo.b[q] = (__bf16)(tmp[q] - (float)hb);
    }
    #pragma unroll
    for (int dt = 0; dt < 4; ++dt) {
      int n = head * DD + dt * 16 + row;
      const __bf16* bph = Wt_hi + (size_t)n * KDIM + k0 + kb;
      const __bf16* bpl = Wt_lo + (size_t)n * KDIM + k0 + kb;
      ABf bh, bl;
      bh.q[0] = *(const uint4*)bph;       bh.q[1] = *(const uint4*)(bph + 16);
      bl.q[0] = *(const uint4*)bpl;       bl.q[1] = *(const uint4*)(bpl + 16);
      acc[dt].v = __builtin_amdgcn_wmma_f32_16x16x32_bf16(false, ahi.v, false, bh.v,
                                                          (short)0, acc[dt].v, false, false);
      acc[dt].v = __builtin_amdgcn_wmma_f32_16x16x32_bf16(false, alo.v, false, bh.v,
                                                          (short)0, acc[dt].v, false, false);
      acc[dt].v = __builtin_amdgcn_wmma_f32_16x16x32_bf16(false, ahi.v, false, bl.v,
                                                          (short)0, acc[dt].v, false, false);
    }
  }

  // fused s1/s2: dot over d of h-row with attention vectors (cross-lane reduce)
  float a1v[4], a2v[4];
  #pragma unroll
  for (int dt = 0; dt < 4; ++dt) {
    a1v[dt] = attn[head * 2 * DD + dt * 16 + row];
    a2v[dt] = attn[head * 2 * DD + DD + dt * 16 + row];
  }
  #pragma unroll
  for (int r = 0; r < 8; ++r) {
    float p1 = 0.f, p2 = 0.f;
    #pragma unroll
    for (int dt = 0; dt < 4; ++dt) { p1 += acc[dt].f[r] * a1v[dt]; p2 += acc[dt].f[r] * a2v[dt]; }
    #pragma unroll
    for (int off = 1; off < 16; off <<= 1) {
      p1 += __shfl_xor(p1, off, 32);
      p2 += __shfl_xor(p2, off, 32);
    }
    if (row == 0) {
      int i = i0 + r + 8 * lh;
      s1t[head * NN + i] = p1;
      s2t[head * NN + i] = p2;
    }
  }

  // store h as bf16 hi + lo residual (row-major [node][head*64+d])
  #pragma unroll
  for (int dt = 0; dt < 4; ++dt)
    #pragma unroll
    for (int r = 0; r < 8; ++r) {
      float f = acc[dt].f[r];
      __bf16 hb = (__bf16)f;
      size_t o = (size_t)(i0 + r + 8 * lh) * OC + head * DD + dt * 16 + row;
      h_hi[o] = hb;
      h_lo[o] = (__bf16)(f - (float)hb);
    }
}

// ---------------- K3: smax2[h] = max_j s2[j,h] (softmax upper bound) ----------------
__global__ void k_smax(const float* __restrict__ s2t, float* __restrict__ smax2) {
  int head = threadIdx.x >> 5, lane = threadIdx.x & 31;
  float m = -3.4e38f;
  for (int j = lane; j < NN; j += 32) m = fmaxf(m, s2t[head * NN + j]);
  #pragma unroll
  for (int off = 1; off < 32; off <<= 1) m = fmaxf(m, __shfl_xor(m, off, 32));
  if (lane == 0) smax2[head] = m;
}

// ---------------- K4: fused masked softmax + P@V aggregation (flash style) ----------
// block = 128 thr = 4 waves; wave w -> head w; block -> 16-row i-tile; j streamed by 32.
// Each wave stages + consumes ONLY its own Vt[head] slice; same-wave DS ops are
// in-order (DScnt), so no s_barrier is needed -- just a compiler scheduling fence.
__global__ void __launch_bounds__(128) k_aggregate(
    const int* __restrict__ adj,
    const __bf16* __restrict__ h_hi, const __bf16* __restrict__ h_lo,
    const float* __restrict__ s1t, const float* __restrict__ s2t,
    const float* __restrict__ smax2, const float* __restrict__ bias,
    float* __restrict__ out) {
  __shared__ __align__(16) __bf16 Vhi[HEADS][DD][40];  // padded: conflict-free b128
  __shared__ __align__(16) __bf16 Vlo[HEADS][DD][40];

  const int lane = threadIdx.x & 31;
  const int head = threadIdx.x >> 5;
  const int row  = lane & 15;
  const int lh   = lane >> 4;
  const int kb   = lh * 8;
  const int klo  = lh * 16;
  const int i0   = blockIdx.x * 16;
  const int i    = i0 + row;

  const float s1v = s1t[head * NN + i];
  const float mhl = leaky(s1v + smax2[head]) * L2E;  // (>= row max) * log2(e)

  Cf acc[4];
  #pragma unroll
  for (int dt = 0; dt < 4; ++dt)
    #pragma unroll
    for (int e = 0; e < 8; ++e) acc[dt].f[e] = 0.0f;
  float ls0 = 0.f, ls1 = 0.f, ls2 = 0.f, ls3 = 0.f;

  // staging role within the wave: 2 j-rows x 32 d-cols each
  const int jj0 = (lane & 15) * 2;
  const int dh  = (lane >> 4) * 32;

  const int*   adjr = adj + (size_t)i * NN;
  const float* s2p  = s2t + head * NN;

  for (int j0 = 0; j0 < NN; j0 += 32) {
    // ---- stage V tile transposed into LDS: Vt[d][j] for this wave's head ----
    {
      const __bf16* r0h = h_hi + (size_t)(j0 + jj0) * OC + head * DD + dh;
      const __bf16* r1h = r0h + OC;
      const __bf16* r0l = h_lo + (size_t)(j0 + jj0) * OC + head * DD + dh;
      const __bf16* r1l = r0l + OC;
      union { uint4 q[4]; unsigned short u[32]; } R0h, R1h, R0l, R1l;
      #pragma unroll
      for (int w = 0; w < 4; ++w) {
        R0h.q[w] = ((const uint4*)r0h)[w];
        R1h.q[w] = ((const uint4*)r1h)[w];
        R0l.q[w] = ((const uint4*)r0l)[w];
        R1l.q[w] = ((const uint4*)r1l)[w];
      }
      #pragma unroll
      for (int d = 0; d < 32; ++d) {
        *(unsigned int*)&Vhi[head][dh + d][jj0] =
            (unsigned int)R0h.u[d] | ((unsigned int)R1h.u[d] << 16);
        *(unsigned int*)&Vlo[head][dh + d][jj0] =
            (unsigned int)R0l.u[d] | ((unsigned int)R1l.u[d] << 16);
      }
    }
    __builtin_amdgcn_wave_barrier();   // keep ds_reads below after ds_writes above

    // ---- scores directly in WMMA A-layout: 16 j's per lane ----
    union { int4 q[4]; int a[16]; } AJ;
    AJ.q[0] = *(const int4*)(adjr + j0 + kb);
    AJ.q[1] = *(const int4*)(adjr + j0 + kb + 4);
    AJ.q[2] = *(const int4*)(adjr + j0 + 16 + kb);
    AJ.q[3] = *(const int4*)(adjr + j0 + 16 + kb + 4);
    union { float4 q[4]; float s[16]; } S2;
    S2.q[0] = *(const float4*)(s2p + j0 + kb);
    S2.q[1] = *(const float4*)(s2p + j0 + kb + 4);
    S2.q[2] = *(const float4*)(s2p + j0 + 16 + kb);
    S2.q[3] = *(const float4*)(s2p + j0 + 16 + kb + 4);
    if (j0 + 128 < NN) __builtin_prefetch(adjr + j0 + 128, 0, 1);  // global_prefetch_b8

    ABf Phi, Plo;
    #pragma unroll
    for (int q = 0; q < 16; ++q) {
      float e = leaky(s1v + S2.s[q]);
      // exp(e - mh) = exp2(e*log2e - mh*log2e); arg <= 0, never overflows
      float p = (AJ.a[q] > 0) ? __builtin_amdgcn_exp2f(__builtin_fmaf(e, L2E, -mhl)) : 0.0f;
      if ((q & 3) == 0) ls0 += p; else if ((q & 3) == 1) ls1 += p;
      else if ((q & 3) == 2) ls2 += p; else ls3 += p;
      __bf16 hb = (__bf16)p;
      Phi.b[q] = hb;
      Plo.b[q] = (__bf16)(p - (float)hb);
    }

    // ---- acc += (Phi+Plo) @ (Vhi+Vlo), dropping the Plo@Vlo term (~2^-17) ----
    #pragma unroll
    for (int dt = 0; dt < 4; ++dt) {
      const __bf16* bph = &Vhi[head][dt * 16 + row][kb];
      const __bf16* bpl = &Vlo[head][dt * 16 + row][kb];
      ABf Bh, Bl;
      Bh.q[0] = *(const uint4*)bph;       Bh.q[1] = *(const uint4*)(bph + 16);
      Bl.q[0] = *(const uint4*)bpl;       Bl.q[1] = *(const uint4*)(bpl + 16);
      acc[dt].v = __builtin_amdgcn_wmma_f32_16x16x32_bf16(false, Phi.v, false, Bh.v,
                                                          (short)0, acc[dt].v, false, false);
      acc[dt].v = __builtin_amdgcn_wmma_f32_16x16x32_bf16(false, Plo.v, false, Bh.v,
                                                          (short)0, acc[dt].v, false, false);
      acc[dt].v = __builtin_amdgcn_wmma_f32_16x16x32_bf16(false, Phi.v, false, Bl.v,
                                                          (short)0, acc[dt].v, false, false);
    }
    __builtin_amdgcn_wave_barrier();   // next iteration's writes stay after these reads
  }

  // lanes L and L+16 hold complementary j-halves of row L&15
  float lsum = (ls0 + ls1) + (ls2 + ls3);
  float lred = lsum + __shfl_xor(lsum, 16, 32);
  lred = fmaxf(lred, 1e-30f);

  #pragma unroll
  for (int r = 0; r < 8; ++r) {
    float lr  = __shfl(lred, r + 8 * lh, 32);   // l for C-layout row of this lane
    float inv = 1.0f / lr;
    #pragma unroll
    for (int dt = 0; dt < 4; ++dt) {
      float b = bias[head * DD + dt * 16 + row];
      out[(size_t)(i0 + r + 8 * lh) * OC + head * DD + dt * 16 + row] =
          acc[dt].f[r] * inv + b;
    }
  }
}

extern "C" void kernel_launch(void* const* d_in, const int* in_sizes, int n_in,
                              void* d_out, int out_size, void* d_ws, size_t ws_size,
                              hipStream_t stream) {
  (void)in_sizes; (void)n_in; (void)out_size; (void)ws_size;
  const float* features  = (const float*)d_in[0];
  const int*   adjacency = (const int*)d_in[1];
  const float* weights   = (const float*)d_in[2];
  const float* attention = (const float*)d_in[3];
  const float* bias      = (const float*)d_in[4];
  float* out = (float*)d_out;

  char* ws = (char*)d_ws;
  __bf16* Wt_hi = (__bf16*)ws;  ws += (size_t)OC * KDIM * 2;
  __bf16* Wt_lo = (__bf16*)ws;  ws += (size_t)OC * KDIM * 2;
  __bf16* h_hi  = (__bf16*)ws;  ws += (size_t)NN * OC * 2;
  __bf16* h_lo  = (__bf16*)ws;  ws += (size_t)NN * OC * 2;
  float* s1t   = (float*)ws;    ws += (size_t)HEADS * NN * 4;
  float* s2t   = (float*)ws;    ws += (size_t)HEADS * NN * 4;
  float* smax2 = (float*)ws;    ws += 64;

  k_prep_w<<<(OC * KDIM + 255) / 256, 256, 0, stream>>>(weights, Wt_hi, Wt_lo);
  k_gemm_h<<<NN / 16, 128, 0, stream>>>(features, attention, Wt_hi, Wt_lo,
                                        h_hi, h_lo, s1t, s2t);
  k_smax<<<1, 128, 0, stream>>>(s2t, smax2);
  k_aggregate<<<NN / 16, 128, 0, stream>>>(adjacency, h_hi, h_lo, s1t, s2t,
                                           smax2, bias, out);
}
